// Decoder_70824010711114
// MI455X (gfx1250) — compile-verified, matrix operand order checked by reading, not run on hardware
//
#include <hip/hip_runtime.h>
#include <hip/hip_bf16.h>
#include <math.h>

// MI455X / gfx1250: wave32, WMMA (not MFMA).
typedef __attribute__((ext_vector_type(16))) __bf16 v16bf;
typedef __attribute__((ext_vector_type(8)))  __bf16 v8bf;
typedef __attribute__((ext_vector_type(8)))  float  v8f;

#define B_    512
#define T_    64
#define S_    512
#define U_    256
#define G3_   768      // 3*UNITS
#define MT    16       // batch tile == WMMA M
#define NT    48       // 768/16 n-tiles
#define KTN   8        // 256/32 k-tiles
#define NWAVE 16
#define NTHR  (NWAVE*32)

// 16-bit operand lane/K mapping (CDNA5 ISA 7.12.2):
// lane L: kbase=(L>>4)*8 ; halves 0..7 -> K=kbase+0..7 ; halves 8..15 -> K=kbase+16..23
__device__ __host__ __forceinline__ int kmap(int lane, int i) {
  return ((lane >> 4) * 8) + (i & 7) + ((i >> 3) * 16);
}

// ---------------- prep: fp32 W/U -> bf16, pre-swizzled into WMMA-B fragment order.
// Fragment for (nt, kt, lane) is 16 contiguous halves: one 32B load in the main loop.
__global__ void prep_weights(const float* __restrict__ W,
                             const float* __restrict__ Umat,
                             __bf16* __restrict__ Wt,
                             __bf16* __restrict__ Ut) {
  int idx = blockIdx.x * blockDim.x + threadIdx.x;      // [0, NT*KTN*32*16)
  if (idx >= NT * KTN * 32 * 16) return;
  int i    = idx & 15;
  int lane = (idx >> 4) & 31;
  int rest = idx >> 9;           // nt*KTN + kt
  int kt   = rest & (KTN - 1);
  int nt   = rest >> 3;
  int n    = nt * 16 + (lane & 15);
  int kl   = kt * 32 + kmap(lane, i);
  if (blockIdx.y == 0) {
    // W rows 1..256 (row 0 = the 'sub' scalar column, handled as a rank-1 update)
    Wt[idx] = (__bf16)W[(size_t)(1 + kl) * G3_ + n];
  } else {
    Ut[idx] = (__bf16)Umat[(size_t)kl * G3_ + n];
  }
}

// ---------------- main: 32 WGs x 16 batch each; full T-loop per WG, no global sync.
__global__ __launch_bounds__(NTHR)
void decoder_kernel(const float* __restrict__ inputs,  // (B,T)
                    const float* __restrict__ hidden,  // (B,U)
                    const float* __restrict__ enc,     // (B,S,U)
                    const float* __restrict__ W,       // (257,768) fp32 (row 0 used here)
                    const float* __restrict__ bias,    // (2,768)
                    const float* __restrict__ dw,      // (256,)
                    const float* __restrict__ db,      // (1,)
                    const __bf16* __restrict__ Wt,     // swizzled bf16
                    const __bf16* __restrict__ Ut,
                    float* __restrict__ out)           // (B,T)
{
  extern __shared__ char smem[];
  float*  s_state = (float*)smem;                 // 16*256 f32  (16 KB)
  float*  s_attn  = s_state + MT * U_;            // 16*512 f32  (32 KB)
  __bf16* s_ctx   = (__bf16*)(s_attn + MT * S_);  // 16*256 bf16 ( 8 KB)
  __bf16* s_h     = s_ctx + MT * U_;              // 16*256 bf16 ( 8 KB)
  float*  s_sub   = (float*)(s_h + MT * U_);      // 16 f32

  const int tid  = threadIdx.x;
  const int lane = tid & 31;
  const int wv   = tid >> 5;               // 0..15 (wave == attention batch row)
  const int bm0  = blockIdx.x * MT;

  // ---- init: state <- hidden, sub <- inputs[:,0]
  for (int idx = tid; idx < MT * U_; idx += NTHR)
    s_state[idx] = hidden[(size_t)(bm0 + (idx >> 8)) * U_ + (idx & 255)];
  if (tid < MT) s_sub[tid] = inputs[(size_t)(bm0 + tid) * T_];
  __syncthreads();

  const int m = wv;
  const float* encb = enc + (size_t)(bm0 + m) * S_ * U_;

  // hoist per-lane loop-invariant weight/bias values (small, cheap to keep live)
  const int   ma  = lane & 15;             // A-frag row
  const int   kb  = (lane >> 4) * 8;       // A-frag K base
  const int   ul  = lane & 15;             // D-frag column (unit offset)
  const int   n0  = wv * 16 + ul;          // unit index 0..255 owned by this (wave,lane)
  const float w0z = W[n0], w0r = W[256 + n0], w0h = W[512 + n0];          // W row 0
  const float b0z = bias[n0],        b0r = bias[256 + n0],        b0h = bias[512 + n0];
  const float b1z = bias[G3_ + n0],  b1r = bias[G3_ + 256 + n0],  b1h = bias[G3_ + 512 + n0];
  float dwl[8];
  #pragma unroll
  for (int j = 0; j < 8; ++j) dwl[j] = dw[lane + 32 * j];
  const float dbv = db[0];

  for (int t = 0; t < T_; ++t) {
    // ---- phase 1: scores[m][s] = state[m] . enc[b,s,:]   (batched matvec, VALU)
    const float* st = s_state + m * U_;
    for (int c = 0; c < S_ / 32; ++c) {
      const int s = lane + 32 * c;
      const float4* ep = (const float4*)(encb + (size_t)s * U_);
      float acc = 0.f;
      #pragma unroll 8
      for (int u4 = 0; u4 < U_ / 4; ++u4) {
        float4 e = ep[u4];
        acc += e.x * st[4 * u4]     + e.y * st[4 * u4 + 1]
             + e.z * st[4 * u4 + 2] + e.w * st[4 * u4 + 3];
      }
      s_attn[m * S_ + s] = acc;
    }
    // ---- phase 2: softmax over S within the wave
    float mx = -1e30f;
    for (int c = 0; c < S_ / 32; ++c) mx = fmaxf(mx, s_attn[m * S_ + lane + 32 * c]);
    #pragma unroll
    for (int o = 16; o; o >>= 1) mx = fmaxf(mx, __shfl_xor(mx, o, 32));
    float sum = 0.f;
    for (int c = 0; c < S_ / 32; ++c) {
      float e = __expf(s_attn[m * S_ + lane + 32 * c] - mx);
      s_attn[m * S_ + lane + 32 * c] = e;
      sum += e;
    }
    #pragma unroll
    for (int o = 16; o; o >>= 1) sum += __shfl_xor(sum, o, 32);
    const float inv = 1.f / sum;

    // ---- phase 3: ctx[m][:] = sum_s attn * enc[b,s,:]  (coalesced over lanes) + bf16 copies
    float acc[8];
    #pragma unroll
    for (int j = 0; j < 8; ++j) acc[j] = 0.f;
    for (int s = 0; s < S_; ++s) {
      const float a = s_attn[m * S_ + s] * inv;
      const float* ep = encb + (size_t)s * U_;
      if (s + 2 < S_) __builtin_prefetch(encb + (size_t)(s + 2) * U_ + lane, 0, 1);
      #pragma unroll
      for (int j = 0; j < 8; ++j) acc[j] += a * ep[lane + 32 * j];
    }
    #pragma unroll
    for (int j = 0; j < 8; ++j) s_ctx[m * U_ + lane + 32 * j] = (__bf16)acc[j];
    #pragma unroll
    for (int j = 0; j < 8; ++j) s_h[m * U_ + lane + 32 * j] = (__bf16)s_state[m * U_ + lane + 32 * j];
    __syncthreads();

    // ---- phase 4: xm = ctx @ W[1:], hm = h @ U  via v_wmma_f32_16x16x32_bf16
    // wave wv owns n-tiles {wv, 16+wv, 32+wv} -> gate columns z/r/h for units [16wv,16wv+16)
    //
    // B fragments are loop-invariant across t; without an opaque token LLVM hoists
    // all 48 of them (384 VGPRs) out of the t-loop and spills to scratch. Laundering
    // the *pointers* through asm downgraded the loads to flat_load (generic aspace,
    // ties up DScnt + aperture checks). Instead, launder a byte OFFSET: the address
    // GEPs stay rooted at the kernarg pointers, so infer-address-spaces keeps the
    // loads as global_load_b128 (SADDR form), while LICM across t is still blocked.
    size_t bofs = 0;
    asm volatile("" : "+s"(bofs));
    const __bf16* Wt_t = Wt + bofs;
    const __bf16* Ut_t = Ut + bofs;

    v8f ax0 = {}, ax1 = {}, ax2 = {};
    v8f ah0 = {}, ah1 = {}, ah2 = {};
    #pragma unroll
    for (int kt = 0; kt < KTN; ++kt) {
      const __bf16* pc = s_ctx + ma * U_ + kt * 32 + kb;
      const __bf16* ph = s_h   + ma * U_ + kt * 32 + kb;
      v16bf a_ctx = __builtin_shufflevector(*(const v8bf*)pc, *(const v8bf*)(pc + 16),
                                            0,1,2,3,4,5,6,7,8,9,10,11,12,13,14,15);
      v16bf a_h   = __builtin_shufflevector(*(const v8bf*)ph, *(const v8bf*)(ph + 16),
                                            0,1,2,3,4,5,6,7,8,9,10,11,12,13,14,15);
      const size_t o0 = ((size_t)((wv)      * KTN + kt) * 32 + lane) * 16;
      const size_t o1 = ((size_t)((wv + 16) * KTN + kt) * 32 + lane) * 16;
      const size_t o2 = ((size_t)((wv + 32) * KTN + kt) * 32 + lane) * 16;
      ax0 = __builtin_amdgcn_wmma_f32_16x16x32_bf16(false, a_ctx, false, *(const v16bf*)(Wt_t + o0), (short)0, ax0, false, false);
      ah0 = __builtin_amdgcn_wmma_f32_16x16x32_bf16(false, a_h,   false, *(const v16bf*)(Ut_t + o0), (short)0, ah0, false, false);
      ax1 = __builtin_amdgcn_wmma_f32_16x16x32_bf16(false, a_ctx, false, *(const v16bf*)(Wt_t + o1), (short)0, ax1, false, false);
      ah1 = __builtin_amdgcn_wmma_f32_16x16x32_bf16(false, a_h,   false, *(const v16bf*)(Ut_t + o1), (short)0, ah1, false, false);
      ax2 = __builtin_amdgcn_wmma_f32_16x16x32_bf16(false, a_ctx, false, *(const v16bf*)(Wt_t + o2), (short)0, ax2, false, false);
      ah2 = __builtin_amdgcn_wmma_f32_16x16x32_bf16(false, a_h,   false, *(const v16bf*)(Ut_t + o2), (short)0, ah2, false, false);
    }

    // ---- phase 5: rank-1 'sub' term + biases + GRU gates, all in accumulator regs.
    // D frag: reg r -> M = r + 8*(lane>>4), column = lane&15.
    #pragma unroll
    for (int r = 0; r < 8; ++r) {
      const int md  = r + 8 * (lane >> 4);
      const float sub = s_sub[md];
      const float xz = ax0[r] + sub * w0z + b0z;
      const float xr = ax1[r] + sub * w0r + b0r;
      const float xh = ax2[r] + sub * w0h + b0h;
      const float hz = ah0[r] + b1z;
      const float hr = ah1[r] + b1r;
      const float hh = ah2[r] + b1h;
      const float z  = 1.f / (1.f + __expf(-(xz + hz)));
      const float rg = 1.f / (1.f + __expf(-(xr + hr)));
      const float c  = tanhf(xh + rg * hh);
      const float hp = s_state[md * U_ + n0];
      s_state[md * U_ + n0] = z * hp + (1.f - z) * c;    // each (md,n0) owned by one lane
    }
    __syncthreads();

    // ---- phase 6: out = relu(state @ dense_w + db); feeds next step's sub
    float p = 0.f;
    #pragma unroll
    for (int j = 0; j < 8; ++j) p += s_state[m * U_ + lane + 32 * j] * dwl[j];
    #pragma unroll
    for (int o = 16; o; o >>= 1) p += __shfl_xor(p, o, 32);
    const float ov = fmaxf(p + dbv, 0.f);
    if (lane == 0) {
      s_sub[m] = ov;
      out[(size_t)(bm0 + m) * T_ + t] = ov;    // ys.T -> (B,T)
    }
    __syncthreads();
  }
}

extern "C" void kernel_launch(void* const* d_in, const int* in_sizes, int n_in,
                              void* d_out, int out_size, void* d_ws, size_t ws_size,
                              hipStream_t stream) {
  const float* inputs = (const float*)d_in[0];
  const float* hidden = (const float*)d_in[1];
  const float* enc    = (const float*)d_in[2];
  const float* W      = (const float*)d_in[3];
  const float* Umat   = (const float*)d_in[4];
  const float* bias   = (const float*)d_in[5];
  const float* dw     = (const float*)d_in[6];
  const float* db     = (const float*)d_in[7];

  __bf16* Wt = (__bf16*)d_ws;                          // 48*8*32*16 halves = 384 KB
  __bf16* Ut = Wt + (size_t)NT * KTN * 32 * 16;        // another 384 KB

  dim3 pg((NT * KTN * 32 * 16 + 255) / 256, 2);
  prep_weights<<<pg, 256, 0, stream>>>(W, Umat, Wt, Ut);

  const size_t shmem = (size_t)MT * U_ * 4 + (size_t)MT * S_ * 4
                     + (size_t)MT * U_ * 2 + (size_t)MT * U_ * 2 + MT * 4;
  decoder_kernel<<<B_ / MT, NTHR, shmem, stream>>>(inputs, hidden, enc, W, bias,
                                                   dw, db, Wt, Ut, (float*)d_out);
}
